// ExportPreQuantizedLayer_50835232916363
// MI455X (gfx1250) — 8-bit matrix operand order repaired and checked by reading, compiled, not run on hardware
//
#include <hip/hip_runtime.h>
#include <hip/hip_bf16.h>

typedef __attribute__((ext_vector_type(8))) int v8i;

#define M_TOTAL 4096
#define N_TOTAL 8192
#define K_TOTAL 2048

__global__ __launch_bounds__(256) void quantize_rows_kernel(
    const float* __restrict__ x, const float* __restrict__ a_scale,
    const int* __restrict__ a_zp,
    unsigned char* __restrict__ xq, int* __restrict__ row_sum)
{
    __shared__ int red[256];
    const int m = blockIdx.x;
    const float inv_s = 1.0f / a_scale[0];
    const float zf = (float)a_zp[0];
    const float* xr = x + (size_t)m * K_TOTAL;
    unsigned char* qr = xq + (size_t)m * K_TOTAL;

    int local = 0;
    #pragma unroll
    for (int k = threadIdx.x; k < K_TOTAL; k += 256) {
        float r = xr[k] * inv_s;
        float v = (r < 0.0f) ? -floorf(-r + 0.5f) : floorf(r + 0.5f);
        float q = fminf(fmaxf(v + zf, 0.0f), 255.0f);
        int qi = (int)q;
        qr[k] = (unsigned char)qi;
        local += qi;
    }
    red[threadIdx.x] = local;
    __syncthreads();
    #pragma unroll
    for (int off = 128; off > 0; off >>= 1) {
        if ((int)threadIdx.x < off) red[threadIdx.x] += red[threadIdx.x + off];
        __syncthreads();
    }
    if (threadIdx.x == 0) row_sum[m] = red[0];
}

__global__ __launch_bounds__(256) void pack_weight_kernel(
    const int* __restrict__ wq, unsigned char* __restrict__ w8,
    int* __restrict__ col_sum)
{
    __shared__ int red[256];
    const int o = blockIdx.x;
    const int* wr = wq + (size_t)o * K_TOTAL;
    unsigned char* pr = w8 + (size_t)o * K_TOTAL;

    int local = 0;
    #pragma unroll
    for (int k = threadIdx.x; k < K_TOTAL; k += 256) {
        int v = wr[k];
        pr[k] = (unsigned char)v;
        local += v;
    }
    red[threadIdx.x] = local;
    __syncthreads();
    #pragma unroll
    for (int off = 128; off > 0; off >>= 1) {
        if ((int)threadIdx.x < off) red[threadIdx.x] += red[threadIdx.x + off];
        __syncthreads();
    }
    if (threadIdx.x == 0) col_sum[o] = red[0];
}

__global__ __launch_bounds__(256) void wmma_iu8_gemm_kernel(
    const unsigned char* __restrict__ Aq,
    const unsigned char* __restrict__ Bq,
    const int* __restrict__ row_sum,
    const int* __restrict__ col_sum,
    const float* __restrict__ a_scale, const int* __restrict__ a_zp,
    const float* __restrict__ w_scale, const int* __restrict__ w_zp,
    const float* __restrict__ bias,
    float* __restrict__ out)
{
    const int lane = threadIdx.x & 31;
    const int wave = threadIdx.x >> 5;
    const int wm = wave >> 2;
    const int wn = wave & 3;

    const int wave_m = blockIdx.y * 128 + wm * 64;
    const int wave_n = blockIdx.x * 128 + wn * 32;

    const int a_m    = lane & 15;
    const int a_koff = (lane >> 4) * 16;
    const int b_n     = lane & 15;
    const int b_kbase = (lane >> 4) * 16;

    const unsigned char* a_base[4];
    #pragma unroll
    for (int i = 0; i < 4; ++i)
        a_base[i] = Aq + (size_t)(wave_m + i * 16 + a_m) * K_TOTAL + a_koff;
    const unsigned char* b_base[2];
    #pragma unroll
    for (int j = 0; j < 2; ++j)
        b_base[j] = Bq + (size_t)(wave_n + j * 16 + b_n) * K_TOTAL + b_kbase;

    v8i acc[4][2];
    #pragma unroll
    for (int i = 0; i < 4; ++i)
        #pragma unroll
        for (int j = 0; j < 2; ++j)
            #pragma unroll
            for (int r = 0; r < 8; ++r)
                acc[i][j][r] = 0;

    for (int k0 = 0; k0 < K_TOTAL; k0 += 64) {
        v8i bfrag[2];
        #pragma unroll
        for (int j = 0; j < 2; ++j) {
            const unsigned char* p = b_base[j] + k0;
            __builtin_prefetch(p + 64, 0, 1);
            const int4 lo = *(const int4*)(p);
            const int4 hi = *(const int4*)(p + 32);
            bfrag[j][0] = lo.x; bfrag[j][1] = lo.y; bfrag[j][2] = lo.z; bfrag[j][3] = lo.w;
            bfrag[j][4] = hi.x; bfrag[j][5] = hi.y; bfrag[j][6] = hi.z; bfrag[j][7] = hi.w;
        }
        #pragma unroll
        for (int i = 0; i < 4; ++i) {
            const unsigned char* p = a_base[i] + k0;
            __builtin_prefetch(p + 64, 0, 1);
            const int2 q0 = *(const int2*)(p);
            const int2 q1 = *(const int2*)(p + 8);
            const int2 q2 = *(const int2*)(p + 32);
            const int2 q3 = *(const int2*)(p + 40);
            v8i afrag;
            afrag[0] = q0.x; afrag[1] = q0.y; afrag[2] = q1.x; afrag[3] = q1.y;
            afrag[4] = q2.x; afrag[5] = q2.y; afrag[6] = q3.x; afrag[7] = q3.y;
            #pragma unroll
            for (int j = 0; j < 2; ++j) {
                acc[i][j] = __builtin_amdgcn_wmma_i32_16x16x64_iu8(
                    false, afrag, false, bfrag[j], acc[i][j], false, false);
            }
        }
    }

    const float aS = a_scale[0];
    const int   aZ = a_zp[0];
    const int c_moff = (lane >> 4) * 8;
    const int c_n    = lane & 15;
    #pragma unroll
    for (int j = 0; j < 2; ++j) {
        const int o = wave_n + j * 16 + c_n;
        const float sc = aS * w_scale[o];
        const int wz = w_zp[o];
        const int corr_o = K_TOTAL * aZ * wz - aZ * col_sum[o];
        const float bs = bias[o];
        #pragma unroll
        for (int i = 0; i < 4; ++i) {
            #pragma unroll
            for (int r = 0; r < 8; ++r) {
                const int m = wave_m + i * 16 + c_moff + r;
                const int v = acc[i][j][r] - wz * row_sum[m] + corr_o;
                out[(size_t)m * N_TOTAL + o] = sc * (float)v + bs;
            }
        }
    }
}

extern "C" void kernel_launch(void* const* d_in, const int* in_sizes, int n_in,
                              void* d_out, int out_size, void* d_ws, size_t ws_size,
                              hipStream_t stream) {
    const float* x       = (const float*)d_in[0];
    const float* a_scale = (const float*)d_in[1];
    const int*   a_zp    = (const int*)d_in[2];
    const int*   weight_q= (const int*)d_in[3];
    const float* w_scale = (const float*)d_in[4];
    const int*   w_zp    = (const int*)d_in[5];
    const float* bias    = (const float*)d_in[6];
    float* out = (float*)d_out;

    char* ws = (char*)d_ws;
    unsigned char* xq = (unsigned char*)ws;
    unsigned char* w8 = (unsigned char*)(ws + (size_t)8 * 1024 * 1024);
    int* row_sum = (int*)(ws + (size_t)24 * 1024 * 1024);
    int* col_sum = (int*)(ws + (size_t)24 * 1024 * 1024 + 16384);

    quantize_rows_kernel<<<M_TOTAL, 256, 0, stream>>>(x, a_scale, a_zp, xq, row_sum);
    pack_weight_kernel<<<N_TOTAL, 256, 0, stream>>>(weight_q, w8, col_sum);

    dim3 grid(N_TOTAL / 128, M_TOTAL / 128);
    wmma_iu8_gemm_kernel<<<grid, 256, 0, stream>>>(
        xq, w8, row_sum, col_sum, a_scale, a_zp, w_scale, w_zp, bias, out);
}
